// TensorAggregateLayer_83459804496132
// MI455X (gfx1250) — compile-verified, matrix operand order checked by reading, not run on hardware
//
#include <hip/hip_runtime.h>
#include <hip/hip_bf16.h>
#include <math.h>

// CDNA5 / gfx1250 fused equivariant tensor-product message passing.
// WMMA (v_wmma_f32_16x16x32_f16) computes the radial filter GEMM
// fn_k = rbf(16x8) @ W_k^T(8x32) per 16-edge tile; the Cartesian
// contraction + atomic scatter runs lane-per-channel afterwards.

typedef __attribute__((ext_vector_type(16))) _Float16 v16h;
typedef __attribute__((ext_vector_type(8)))  float    v8f;

#define C_CH            32
#define NMAX            8
#define CUTOFF_F        5.0f
#define INV_NORM        0.03125f     // 1/32 folded into filters
#define WAVES_PER_BLOCK 8
#define TILES_PER_WAVE  4
#define EDGES_PER_TILE  16
#define EDGES_PER_WAVE  (TILES_PER_WAVE * EDGES_PER_TILE)   // 64
#define EDGES_PER_BLOCK (WAVES_PER_BLOCK * EDGES_PER_WAVE)  // 512

__global__ void tp_zero_kernel(float* __restrict__ p, int n) {
    int i = blockIdx.x * blockDim.x + threadIdx.x;
    if (i < n) p[i] = 0.0f;
}

__launch_bounds__(256)
__global__ void tp_edge_kernel(const float* __restrict__ T0,
                               const float* __restrict__ T1,
                               const float* __restrict__ T2,
                               const float* __restrict__ W0,
                               const float* __restrict__ W1,
                               const float* __restrict__ W2,
                               const float* __restrict__ coord,
                               const int*   __restrict__ eidx,   // (2, E) flat
                               float* __restrict__ out0,
                               float* __restrict__ out1,
                               float* __restrict__ out2,
                               int n_edges)
{
    // Per-wave LDS region: fn[3][16][32] floats + u[16][4] floats = 1600 floats.
    __shared__ float lds[WAVES_PER_BLOCK][1600];

    const int tid  = threadIdx.x;
    const int wave = tid >> 5;
    const int lane = tid & 31;
    const int l16  = lane & 15;
    float* __restrict__ fn_lds = &lds[wave][0];     // [k][edge][chan]
    float* __restrict__ u_lds  = &lds[wave][1536];  // [edge][4]

    // ---- Build WMMA B operands once: B[k][n0] = W_k^T column tile ----
    // B is K(=32, real 8) x N(=16). Lane n (0..15) holds column
    // channel = n0*16+n, halves 0..7 = K 0..7; everything else zero-padded.
    v16h B[3][2];
    #pragma unroll
    for (int k = 0; k < 3; ++k) {
        const float* __restrict__ W = (k == 0) ? W0 : (k == 1) ? W1 : W2;
        #pragma unroll
        for (int n0 = 0; n0 < 2; ++n0) {
            v16h b = {};
            if (lane < 16) {
                const int ch = n0 * 16 + l16;
                #pragma unroll
                for (int q = 0; q < NMAX; ++q)
                    b[q] = (_Float16)W[ch * NMAX + q];
            }
            B[k][n0] = b;
        }
    }

    const long wave_base =
        ((long)blockIdx.x * WAVES_PER_BLOCK + wave) * EDGES_PER_WAVE;

    for (int t2 = 0; t2 < TILES_PER_WAVE; ++t2) {
        const long tile_base = wave_base + (long)t2 * EDGES_PER_TILE;

        // ================= Stage 1: geometry + radial basis =================
        // Edge for this lane's A-row (lanes 16..31 mirror lanes 0..15).
        const long e = tile_base + l16;
        v16h A = {};
        if (e < n_edges) {
            const int ai = eidx[e];
            const int aj = eidx[n_edges + e];
            const float rx = coord[3*aj+0] - coord[3*ai+0];
            const float ry = coord[3*aj+1] - coord[3*ai+1];
            const float rz = coord[3*aj+2] - coord[3*ai+2];
            const float d2 = rx*rx + ry*ry + rz*rz + 1e-12f;
            const float d  = sqrtf(d2);
            const float inv_d = 1.0f / d;

            // arg = pi*d/cutoff; cutoff fc uses cos(arg) directly.
            const float arg = (float)M_PI * d * (1.0f / CUTOFF_F);
            float s1, c1;
            __sincosf(arg, &s1, &c1);
            const float fc = (d < CUTOFF_F) ? 0.5f * (c1 + 1.0f) : 0.0f;
            // sqrt(2/cutoff) * (1/d) * fc * (1/NORM) folded into all basis vals
            const float pref = 0.632455532f * inv_d * fc * INV_NORM;

            if (lane < 16) {
                // u stored for stage 2
                u_lds[l16 * 4 + 0] = rx * inv_d;
                u_lds[l16 * 4 + 1] = ry * inv_d;
                u_lds[l16 * 4 + 2] = rz * inv_d;
                // Chebyshev-style recurrence: sin((n+1)a)=2c1*sin(na)-sin((n-1)a)
                float sp = 0.0f, sc = s1;
                const float twoc = 2.0f * c1;
                #pragma unroll
                for (int n = 0; n < NMAX; ++n) {
                    A[n] = (_Float16)(pref * sc);
                    const float sn = twoc * sc - sp;
                    sp = sc; sc = sn;
                }
            }
        }

        // ================= WMMA filter GEMM (uniform control flow) ==========
        // D(16x16) = A(16x32,f16) x B(32x16,f16); 2 column tiles x 3 weights.
        const int mrow = (lane >= 16) ? 8 : 0;  // C/D layout: VGPR r -> row r(+8)
        #pragma unroll
        for (int k = 0; k < 3; ++k) {
            #pragma unroll
            for (int n0 = 0; n0 < 2; ++n0) {
                v8f dacc = {};
                dacc = __builtin_amdgcn_wmma_f32_16x16x32_f16(
                    /*neg_a=*/false, A, /*neg_b=*/false, B[k][n0],
                    /*c_mod=*/(short)0, dacc,
                    /*reuse_a=*/false, /*reuse_b=*/false);
                const int ncol = n0 * 16 + l16;
                #pragma unroll
                for (int r = 0; r < 8; ++r)
                    fn_lds[k * 512 + (mrow + r) * 32 + ncol] = dacc[r];
            }
        }

        __syncthreads();   // publish fn/u (uniform barrier count per block)

        // ================= Stage 2: contraction + atomic scatter ============
        // lane = channel; loop over the 16 edges of this tile.
        const int c = lane;
        for (int t = 0; t < EDGES_PER_TILE; ++t) {
            const long e2 = tile_base + t;
            if (e2 >= n_edges) break;   // wave-uniform
            const int ai = eidx[e2];
            const int aj = eidx[n_edges + e2];
            const float ux = u_lds[t*4+0];
            const float uy = u_lds[t*4+1];
            const float uz = u_lds[t*4+2];
            const float f0 = fn_lds[0*512 + t*32 + c];
            const float f1 = fn_lds[1*512 + t*32 + c];
            const float f2 = fn_lds[2*512 + t*32 + c];

            const long jc = (long)aj * C_CH + c;
            const float s0 = T0[jc];
            const float* __restrict__ vp = T1 + jc * 3;
            const float* __restrict__ Mp = T2 + jc * 9;
            const float v0 = vp[0], v1 = vp[1], v2 = vp[2];
            const float m00 = Mp[0], m01 = Mp[1], m02 = Mp[2];
            const float m10 = Mp[3], m11 = Mp[4], m12 = Mp[5];
            const float m20 = Mp[6], m21 = Mp[7], m22 = Mp[8];

            const float vu  = v0*ux + v1*uy + v2*uz;
            const float Mu0 = m00*ux + m01*uy + m02*uz;
            const float Mu1 = m10*ux + m11*uy + m12*uz;
            const float Mu2 = m20*ux + m21*uy + m22*uz;
            const float uMu = ux*Mu0 + uy*Mu1 + uz*Mu2;

            const long ic = (long)ai * C_CH + c;

            // order-0: s0*f0 + f1*(v.u) + f2*(u.M.u)
            atomicAdd(out0 + ic, fmaf(s0, f0, fmaf(f1, vu, f2 * uMu)));

            // order-1: f0*v + (f1*s0 + f2*(v.u))*u + f1*(M u)
            const float g = fmaf(f1, s0, f2 * vu);
            float* __restrict__ o1 = out1 + ic * 3;
            atomicAdd(o1 + 0, fmaf(f0, v0, fmaf(g, ux, f1 * Mu0)));
            atomicAdd(o1 + 1, fmaf(f0, v1, fmaf(g, uy, f1 * Mu1)));
            atomicAdd(o1 + 2, fmaf(f0, v2, fmaf(g, uz, f1 * Mu2)));

            // order-2: f0*M + (f2*s0*u[a] + f1*v[a] + f2*Mu[a]) * u[b]
            const float f2s0 = f2 * s0;
            const float h0 = fmaf(f2s0, ux, fmaf(f1, v0, f2 * Mu0));
            const float h1 = fmaf(f2s0, uy, fmaf(f1, v1, f2 * Mu1));
            const float h2 = fmaf(f2s0, uz, fmaf(f1, v2, f2 * Mu2));
            float* __restrict__ o2 = out2 + ic * 9;
            atomicAdd(o2 + 0, fmaf(f0, m00, h0 * ux));
            atomicAdd(o2 + 1, fmaf(f0, m01, h0 * uy));
            atomicAdd(o2 + 2, fmaf(f0, m02, h0 * uz));
            atomicAdd(o2 + 3, fmaf(f0, m10, h1 * ux));
            atomicAdd(o2 + 4, fmaf(f0, m11, h1 * uy));
            atomicAdd(o2 + 5, fmaf(f0, m12, h1 * uz));
            atomicAdd(o2 + 6, fmaf(f0, m20, h2 * ux));
            atomicAdd(o2 + 7, fmaf(f0, m21, h2 * uy));
            atomicAdd(o2 + 8, fmaf(f0, m22, h2 * uz));
        }

        __syncthreads();   // before next tile overwrites this wave's LDS
    }
}

extern "C" void kernel_launch(void* const* d_in, const int* in_sizes, int n_in,
                              void* d_out, int out_size, void* d_ws, size_t ws_size,
                              hipStream_t stream) {
    const float* T0    = (const float*)d_in[0];
    const float* T1    = (const float*)d_in[1];
    const float* T2    = (const float*)d_in[2];
    const float* W0    = (const float*)d_in[3];
    const float* W1    = (const float*)d_in[4];
    const float* W2    = (const float*)d_in[5];
    const float* coord = (const float*)d_in[6];
    const int*   eidx  = (const int*)d_in[7];
    // d_in[8] (atomic_number) only defines n_atoms.

    const int n_edges = in_sizes[7] / 2;
    const int n_atoms = in_sizes[8];

    float* out  = (float*)d_out;
    float* out0 = out;                              // (N, 32)
    float* out1 = out + (size_t)n_atoms * 32;       // (N, 32, 3)
    float* out2 = out + (size_t)n_atoms * 32 * 4;   // (N, 32, 3, 3)

    // Zero the (poisoned) output; we accumulate with atomics.
    tp_zero_kernel<<<(out_size + 255) / 256, 256, 0, stream>>>(out, out_size);

    const int blocks = (n_edges + EDGES_PER_BLOCK - 1) / EDGES_PER_BLOCK;
    tp_edge_kernel<<<blocks, 256, 0, stream>>>(T0, T1, T2, W0, W1, W2,
                                               coord, eidx,
                                               out0, out1, out2, n_edges);
}
